// KVCacheLightweight_87101936763221
// MI455X (gfx1250) — matrix operand order repaired and checked by reading, not compiled, not run on hardware
//
#include <hip/hip_runtime.h>
#include <hip/hip_bf16.h>

typedef __attribute__((ext_vector_type(16))) _Float16 v16h;
typedef __attribute__((ext_vector_type(8)))  float    v8f;

#define GLOBAL_TOKENS 4
#define QMAXF 15.0f

__device__ __forceinline__ unsigned sort_key(float f) {
    unsigned b = __float_as_uint(f);
    return (b & 0x80000000u) ? ~b : (b | 0x80000000u);
}

// ---------------- K0: init workspace argmin slots ----------------
__global__ void kvlw_init(unsigned long long* am, int H) {
    int i = threadIdx.x;
    if (i < H) am[i] = 0xFFFFFFFFFFFFFFFFull;
}

// ---------------- K1: k streaming dequant/requant + WMMA scorer ----------------
// grid = (L/128, H), block = 256 (8 waves); each wave owns a 16-token tile.
__global__ __launch_bounds__(256) void kvlw_k_pass(
    const int* __restrict__ kq, const float* __restrict__ ksc, const float* __restrict__ kz,
    const int* __restrict__ pos, const float* __restrict__ w,
    float* __restrict__ out_kv_k, float* __restrict__ out_kq,
    float* __restrict__ out_ksc, float* __restrict__ out_kz,
    unsigned long long* __restrict__ argmin_ws, int H, int L)
{
    const int D = 128;
    __shared__ float sw[128];
    __shared__ unsigned long long cand[16];

    const int h    = blockIdx.y;
    const int tid  = threadIdx.x;
    const int wid  = tid >> 5;
    const int lane = tid & 31;

    if (tid < 128) sw[tid] = w[(size_t)h * 128 + tid];
    __syncthreads();

    const int l0 = blockIdx.x * 128 + wid * 16;

    // ---- Phase A: coalesced stream: dequant -> kv, row min/max -> requant ----
    for (int r = 0; r < 16; ++r) {
        const int l = l0 + r;
        const size_t tok = (size_t)h * L + l;
        const size_t rowoff = tok * D;
        int4 q = ((const int4*)(kq + rowoff))[lane];
        const float s = ksc[tok];
        const float z = kz[tok];
        float x0 = (float)q.x * s + z;
        float x1 = (float)q.y * s + z;
        float x2 = (float)q.z * s + z;
        float x3 = (float)q.w * s + z;
        float4 xv; xv.x = x0; xv.y = x1; xv.z = x2; xv.w = x3;
        ((float4*)(out_kv_k + rowoff))[lane] = xv;

        float mn = fminf(fminf(x0, x1), fminf(x2, x3));
        float mx = fmaxf(fmaxf(x0, x1), fmaxf(x2, x3));
        #pragma unroll
        for (int m = 1; m < 32; m <<= 1) {
            mn = fminf(mn, __shfl_xor(mn, m, 32));
            mx = fmaxf(mx, __shfl_xor(mx, m, 32));
        }
        const float sc  = fmaxf((mx - mn) * (1.0f / QMAXF), 1e-6f);
        const float inv = 1.0f / sc;
        float4 qn;
        qn.x = fminf(fmaxf(rintf((x0 - mn) * inv), 0.0f), QMAXF);
        qn.y = fminf(fmaxf(rintf((x1 - mn) * inv), 0.0f), QMAXF);
        qn.z = fminf(fmaxf(rintf((x2 - mn) * inv), 0.0f), QMAXF);
        qn.w = fminf(fmaxf(rintf((x3 - mn) * inv), 0.0f), QMAXF);
        ((float4*)(out_kq + rowoff))[lane] = qn;
        if (lane == 0) { out_ksc[tok] = sc; out_kz[tok] = mn; }
    }

    // ---- Phase B: WMMA scorer (scores = A(16x128) . w, B replicated over N) ----
    const int row  = lane & 15;
    const int half = lane >> 4;
    const int la   = l0 + row;
    const size_t atok = (size_t)h * L + la;
    const size_t arow = atok * D;
    const float as = ksc[atok];
    const float az = kz[atok];

    v8f c = {0.f, 0.f, 0.f, 0.f, 0.f, 0.f, 0.f, 0.f};
    #pragma unroll
    for (int ch = 0; ch < 4; ++ch) {
        const int b1 = ch * 32 + half * 8;     // lane's first K segment
        int4 qa = *(const int4*)(kq + arow + b1);
        int4 qb = *(const int4*)(kq + arow + b1 + 4);
        int4 qc = *(const int4*)(kq + arow + b1 + 16);
        int4 qd = *(const int4*)(kq + arow + b1 + 20);
        v16h a;
        a[0]  = (_Float16)((float)qa.x * as + az);
        a[1]  = (_Float16)((float)qa.y * as + az);
        a[2]  = (_Float16)((float)qa.z * as + az);
        a[3]  = (_Float16)((float)qa.w * as + az);
        a[4]  = (_Float16)((float)qb.x * as + az);
        a[5]  = (_Float16)((float)qb.y * as + az);
        a[6]  = (_Float16)((float)qb.z * as + az);
        a[7]  = (_Float16)((float)qb.w * as + az);
        a[8]  = (_Float16)((float)qc.x * as + az);
        a[9]  = (_Float16)((float)qc.y * as + az);
        a[10] = (_Float16)((float)qc.z * as + az);
        a[11] = (_Float16)((float)qc.w * as + az);
        a[12] = (_Float16)((float)qd.x * as + az);
        a[13] = (_Float16)((float)qd.y * as + az);
        a[14] = (_Float16)((float)qd.z * as + az);
        a[15] = (_Float16)((float)qd.w * as + az);
        const int wb = ch * 32 + half * 8;     // K rows: wb..wb+7 and wb+16..wb+23
        v16h b;
        #pragma unroll
        for (int j = 0; j < 16; ++j) b[j] = (_Float16)sw[wb + (j & 7) + ((j >> 3) << 4)];
        c = __builtin_amdgcn_wmma_f32_16x16x32_f16(
                false, a, false, b, (short)0, c, false, false);
    }

    // lane<16 holds tokens M=0..7 in c[0..7]; lane>=16 holds M=8..15
    unsigned long long best = 0xFFFFFFFFFFFFFFFFull;
    #pragma unroll
    for (int m = 0; m < 8; ++m) {
        const int ll = l0 + half * 8 + m;
        const int p  = pos[(size_t)h * L + ll];
        unsigned key;
        if (p == -1)                 key = 0x007FFFFFu;  // sort_key(-inf): wins
        else if (ll < GLOBAL_TOKENS) key = 0xFF800000u;  // sort_key(+inf): protected
        else                         key = sort_key(c[m]);
        unsigned long long pk = ((unsigned long long)key << 32) | (unsigned)ll;
        if (pk < best) best = pk;
    }
    if ((lane & 15) == 0) cand[wid * 2 + half] = best;
    __syncthreads();
    if (tid == 0) {
        unsigned long long b = cand[0];
        #pragma unroll
        for (int i = 1; i < 16; ++i) if (cand[i] < b) b = cand[i];
        atomicMin(&argmin_ws[h], b);
    }
}

// ---------------- K2: finalize eviction per head ----------------
__global__ void kvlw_finalize(const unsigned long long* __restrict__ am,
                              const int* __restrict__ pos,
                              int* __restrict__ evict, float* __restrict__ out_num,
                              int H, int L)
{
    int h = threadIdx.x;
    if (h >= H) return;
    int e = (int)(unsigned)am[h];
    evict[h] = e;
    out_num[h] = (pos[(size_t)h * L + e] == -1) ? 1.0f : 0.0f;
}

// ---------------- K3a: pos_new stream ----------------
__global__ void kvlw_pos_pass(const int* __restrict__ pos, const int* __restrict__ evict,
                              const int* __restrict__ input_pos,
                              float* __restrict__ out_pos, int H, int L)
{
    int i = blockIdx.x * blockDim.x + threadIdx.x;
    if (i >= H * L) return;
    int h = i / L, l = i - h * L;
    int p = (l == evict[h]) ? *input_pos : pos[i];
    out_pos[i] = (float)p;
}

// ---------------- K3b: rewrite evicted k rows from k_val ----------------
__global__ __launch_bounds__(32) void kvlw_k_fixup(
    const float* __restrict__ kval, const int* __restrict__ evict,
    float* __restrict__ out_kv_k, float* __restrict__ out_kq,
    float* __restrict__ out_ksc, float* __restrict__ out_kz, int L)
{
    const int h = blockIdx.x, lane = threadIdx.x;
    const int e = evict[h];
    float4 x = ((const float4*)(kval + (size_t)h * 128))[lane];
    const size_t tok = (size_t)h * L + e;
    const size_t rowoff = tok * 128;
    ((float4*)(out_kv_k + rowoff))[lane] = x;
    float mn = fminf(fminf(x.x, x.y), fminf(x.z, x.w));
    float mx = fmaxf(fmaxf(x.x, x.y), fmaxf(x.z, x.w));
    #pragma unroll
    for (int m = 1; m < 32; m <<= 1) {
        mn = fminf(mn, __shfl_xor(mn, m, 32));
        mx = fmaxf(mx, __shfl_xor(mx, m, 32));
    }
    const float sc  = fmaxf((mx - mn) * (1.0f / QMAXF), 1e-6f);
    const float inv = 1.0f / sc;
    float4 qn;
    qn.x = fminf(fmaxf(rintf((x.x - mn) * inv), 0.0f), QMAXF);
    qn.y = fminf(fmaxf(rintf((x.y - mn) * inv), 0.0f), QMAXF);
    qn.z = fminf(fmaxf(rintf((x.z - mn) * inv), 0.0f), QMAXF);
    qn.w = fminf(fmaxf(rintf((x.w - mn) * inv), 0.0f), QMAXF);
    ((float4*)(out_kq + rowoff))[lane] = qn;
    if (lane == 0) { out_ksc[tok] = sc; out_kz[tok] = mn; }
}

// ---------------- K4: v stream with inline fill ----------------
// grid = (L/8, H), block = 256 (8 waves); one wave per token row.
__global__ __launch_bounds__(256) void kvlw_v_pass(
    const int* __restrict__ vq, const float* __restrict__ vsc, const float* __restrict__ vz,
    const float* __restrict__ vval, const int* __restrict__ evict,
    float* __restrict__ out_kv_v, float* __restrict__ out_vq,
    float* __restrict__ out_vsc, float* __restrict__ out_vz, int H, int L)
{
    const int h    = blockIdx.y;
    const int wid  = threadIdx.x >> 5;
    const int lane = threadIdx.x & 31;
    const int l    = blockIdx.x * 8 + wid;
    const size_t tok = (size_t)h * L + l;
    const size_t rowoff = tok * 128;

    float4 x;
    if (l == evict[h]) {  // wave-uniform branch
        x = ((const float4*)(vval + (size_t)h * 128))[lane];
    } else {
        int4 q = ((const int4*)(vq + rowoff))[lane];
        const float s = vsc[tok], z = vz[tok];
        x.x = (float)q.x * s + z; x.y = (float)q.y * s + z;
        x.z = (float)q.z * s + z; x.w = (float)q.w * s + z;
    }
    ((float4*)(out_kv_v + rowoff))[lane] = x;

    float mn = fminf(fminf(x.x, x.y), fminf(x.z, x.w));
    float mx = fmaxf(fmaxf(x.x, x.y), fmaxf(x.z, x.w));
    #pragma unroll
    for (int m = 1; m < 32; m <<= 1) {
        mn = fminf(mn, __shfl_xor(mn, m, 32));
        mx = fmaxf(mx, __shfl_xor(mx, m, 32));
    }
    const float sc  = fmaxf((mx - mn) * (1.0f / QMAXF), 1e-6f);
    const float inv = 1.0f / sc;
    float4 qn;
    qn.x = fminf(fmaxf(rintf((x.x - mn) * inv), 0.0f), QMAXF);
    qn.y = fminf(fmaxf(rintf((x.y - mn) * inv), 0.0f), QMAXF);
    qn.z = fminf(fmaxf(rintf((x.z - mn) * inv), 0.0f), QMAXF);
    qn.w = fminf(fmaxf(rintf((x.w - mn) * inv), 0.0f), QMAXF);
    ((float4*)(out_vq + rowoff))[lane] = qn;
    if (lane == 0) { out_vsc[tok] = sc; out_vz[tok] = mn; }
}

extern "C" void kernel_launch(void* const* d_in, const int* in_sizes, int n_in,
                              void* d_out, int out_size, void* d_ws, size_t ws_size,
                              hipStream_t stream) {
    const int*   kq        = (const int*)  d_in[0];
    const int*   vq        = (const int*)  d_in[1];
    const float* ksc       = (const float*)d_in[2];
    const float* kz        = (const float*)d_in[3];
    const float* vsc       = (const float*)d_in[4];
    const float* vz        = (const float*)d_in[5];
    const int*   pos       = (const int*)  d_in[6];
    const int*   input_pos = (const int*)  d_in[7];
    const float* kval      = (const float*)d_in[8];
    const float* vval      = (const float*)d_in[9];
    const float* w         = (const float*)d_in[10];

    // Derive shapes (B=1): D = HLD/HL; H = (H*D)/D; L = HL/H
    const long long HLD = in_sizes[0];
    const long long HL  = in_sizes[2];
    const int D = (int)(HLD / HL);           // 128
    const int H = in_sizes[8] / D;           // 32
    const int L = (int)(HL / H);             // 8192
    (void)D;

    float* out      = (float*)d_out;
    float* out_kv_k = out;
    float* out_kv_v = out + (size_t)HL * 128;
    float* out_num  = out + 2 * (size_t)HL * 128;
    float* out_pos  = out_num + H;
    float* out_kq   = out_pos + HL;
    float* out_ksc  = out_kq + (size_t)HL * 128;
    float* out_kz   = out_ksc + HL;
    float* out_vq   = out_kz + HL;
    float* out_vsc  = out_vq + (size_t)HL * 128;
    float* out_vz   = out_vsc + HL;

    unsigned long long* am = (unsigned long long*)d_ws;
    int* evict = (int*)(am + H);

    kvlw_init<<<1, 64, 0, stream>>>(am, H);
    kvlw_k_pass<<<dim3(L / 128, H), 256, 0, stream>>>(
        kq, ksc, kz, pos, w, out_kv_k, out_kq, out_ksc, out_kz, am, H, L);
    kvlw_finalize<<<1, 64, 0, stream>>>(am, pos, evict, out_num, H, L);
    kvlw_pos_pass<<<(H * L + 255) / 256, 256, 0, stream>>>(
        pos, evict, input_pos, out_pos, H, L);
    kvlw_k_fixup<<<H, 32, 0, stream>>>(
        kval, evict, out_kv_k, out_kq, out_ksc, out_kz, L);
    kvlw_v_pass<<<dim3(L / 8, H), 256, 0, stream>>>(
        vq, vsc, vz, vval, evict, out_kv_v, out_vq, out_vsc, out_vz, H, L);
}
